// Generator_17824114278765
// MI455X (gfx1250) — matrix operand order repaired and checked by reading, not compiled, not run on hardware
//
#include <hip/hip_runtime.h>
#include <hip/hip_bf16.h>

// ---------------------------------------------------------------------------
// MI455X (gfx1250). Score GEMM (4096x4096x576/batch) and deconv GEMM
// (4096x1024x4096/batch) run on v_wmma_f32_16x16x32_bf16 with bf16 hi/lo
// splitting (3 WMMAs per fp32 product ~= fp32 accuracy). All GEMM operands are
// pre-packed to bf16 hi/lo planes so the K-loop staging is pure contiguous
// copies -> GLOBAL_LOAD_ASYNC_TO_LDS_B64 (ASYNCcnt) with double-buffered LDS.
// Working set ~126MB, L2-resident pieces per batch; compute bound (~240 GFLOP
// fp32-equivalent) => matrix pipe is the right target.
// ---------------------------------------------------------------------------

constexpr int B_ = 4, C_ = 64, H_ = 128, W_ = 128;
constexpr int Hs = 64, Ws = 64;          // fg after /2 resize
constexpr int P_ = Hs * Ws;              // 4096 patches == 4096 pixels
constexpr int K1 = C_ * 9;               // 576  (score GEMM contraction)
constexpr int K2 = C_ * 16;              // 1024 (deconv GEMM N-dim: c*16+kh*4+kw)

typedef __attribute__((ext_vector_type(16))) __bf16 v16bf;
typedef __attribute__((ext_vector_type(8)))  float  v8f;

#define BM 128
#define BN 64
#define BK 32
#define LDK 36   // ushort row stride: 72B -> 8B-aligned rows, 18-dword stride
                 // (18*k mod 64 distinct for k<16 => conflict-free frag reads)

__device__ __forceinline__ unsigned short f2bf_hi(float f) {
    unsigned int u = __builtin_bit_cast(unsigned int, f);
    unsigned int r = u + 0x7FFFu + ((u >> 16) & 1u);   // round-to-nearest-even
    return (unsigned short)(r >> 16);
}
__device__ __forceinline__ float bf2f(unsigned short s) {
    unsigned int u = ((unsigned int)s) << 16;
    return __builtin_bit_cast(float, u);
}

// ---- CDNA5 async global->LDS path (guarded; plain copy fallback) ----------
// Builtin signature (from hipcc diagnostic): param0 = v2i* in AS1 (global src),
// param1 = v2i* in AS3 (LDS dst), then imm offset, imm cpol.
#if __has_builtin(__builtin_amdgcn_global_load_async_to_lds_b64)
#define HAS_ASYNC_LDS 1
typedef int v2i_ __attribute__((vector_size(8)));        // generic vector kind
typedef __attribute__((address_space(1))) v2i_ as1_v2i;  // global
typedef __attribute__((address_space(3))) v2i_ as3_v2i;  // LDS
#endif

__device__ __forceinline__ void cp8(void* lds, const void* g) {
#ifdef HAS_ASYNC_LDS
    __builtin_amdgcn_global_load_async_to_lds_b64((as1_v2i*)g, (as3_v2i*)lds, 0, 0);
#else
    *(uint2*)lds = *(const uint2*)g;
#endif
}
template <int N>
__device__ __forceinline__ void async_wait() {
#ifdef HAS_ASYNC_LDS
#if __has_builtin(__builtin_amdgcn_s_wait_asynccnt)
    __builtin_amdgcn_s_wait_asynccnt(N);
#else
    asm volatile("s_wait_asynccnt %0" ::"n"(N));
#endif
#endif
}

union Frag { v16bf v; unsigned int x[8]; };

// A fragment (16x32 bf16), LDS row-major [m][k]:
// lanes 0-15: M=lane, K 0..7 & 16..23; lanes 16-31: same M, K 8..15 & 24..31.
__device__ __forceinline__ void load_afrag(Frag& f, const unsigned short (*S)[LDK],
                                           int rowBase, int l, int h16) {
    const unsigned int* p = (const unsigned int*)S[rowBase + l];
#pragma unroll
    for (int q = 0; q < 4; ++q) {
        f.x[q]     = p[h16 * 4 + q];
        f.x[4 + q] = p[8 + h16 * 4 + q];
    }
}
// B fragment (32x16 bf16) from LDS holding B transposed [n][k]:
// lane's column N = lane&15; lanes 0-15: K 0..7 & 16..23; lanes 16-31: K 8..15 & 24..31.
__device__ __forceinline__ void load_bfrag(Frag& f, const unsigned short (*S)[LDK],
                                           int rowBase, int l, int h16) {
    const unsigned int* p = (const unsigned int*)S[rowBase + l];
#pragma unroll
    for (int q = 0; q < 4; ++q) {
        f.x[q]     = p[h16 * 4 + q];
        f.x[4 + q] = p[8 + h16 * 4 + q];
    }
}

#define WMMA_BF16(A, Bx, Cx) \
    __builtin_amdgcn_wmma_f32_16x16x32_bf16(false, (A), false, (Bx), (short)0, (Cx), false, false)

// ---------------------------------------------------------------------------
// Unified GEMM: out[r,c] = scale(c) * sum_k A[r,k]*B[c,k], hi/lo bf16 planes.
// Operand rows: [hi: K ushorts][lo: K ushorts], row stride = 4*K bytes.
// Double-buffered LDS, async staging (12 wave-ops per stage).
// ---------------------------------------------------------------------------
template <int KTOT, bool COLSCALE>
__global__ void __launch_bounds__(256) gemm_kernel(const char* __restrict__ Ab, int arow,
                                                   const char* __restrict__ Bb, int brow,
                                                   float* __restrict__ out, int ldo,
                                                   const float* __restrict__ cs) {
    struct alignas(16) Tiles {
        unsigned short Ah[BM][LDK], Al[BM][LDK];
        unsigned short Bh[BN][LDK], Bl[BN][LDK];
    };
    __shared__ Tiles tl[2];

    const int tid = threadIdx.x;
    const int lane = tid & 31, wv = tid >> 5;
    const int wm = wv & 3, wn = wv >> 2;          // 4x2 wave tiling of 128x64
    const int l = lane & 15, h16 = lane >> 4;
    const int R0 = blockIdx.x * BM, C0 = blockIdx.y * BN;
    const int ahalf = arow >> 1, bhalf = brow >> 1;

    auto stage = [&](Tiles& T, int k0) {
        const int koff = k0 << 1;
#pragma unroll
        for (int i = 0; i < 8; ++i) {              // A: 128 rows x (64B hi + 64B lo)
            int s = i * 256 + tid;
            int ml = s >> 4, t = s & 15, pl = t >> 3, bo = (t & 7) << 3;
            char* dst = (char*)(pl ? T.Al[ml] : T.Ah[ml]) + bo;
            const char* src = Ab + (size_t)(R0 + ml) * arow + pl * ahalf + koff + bo;
            cp8(dst, src);
        }
#pragma unroll
        for (int i = 0; i < 4; ++i) {              // B: 64 rows x (64B hi + 64B lo)
            int s = i * 256 + tid;
            int nl = s >> 4, t = s & 15, pl = t >> 3, bo = (t & 7) << 3;
            char* dst = (char*)(pl ? T.Bl[nl] : T.Bh[nl]) + bo;
            const char* src = Bb + (size_t)(C0 + nl) * brow + pl * bhalf + koff + bo;
            cp8(dst, src);
        }
    };

    v8f acc[2][2];
#pragma unroll
    for (int i = 0; i < 2; ++i)
#pragma unroll
        for (int j = 0; j < 2; ++j) acc[i][j] = (v8f)0.0f;

    constexpr int NIT = KTOT / BK;
    stage(tl[0], 0);
    for (int it = 0; it < NIT; ++it) {
        Tiles& cur = tl[it & 1];
        if (it + 1 < NIT) {
            stage(tl[(it + 1) & 1], (it + 1) * BK);  // prefetch next stage
            async_wait<12>();                        // retire stage 'it' (in-order)
        } else {
            async_wait<0>();
        }
        __syncthreads();                             // stage 'it' visible to all waves

        Frag ah[2], al[2], bh[2], bl[2];
#pragma unroll
        for (int i = 0; i < 2; ++i) {
            load_afrag(ah[i], cur.Ah, wm * 32 + i * 16, l, h16);
            load_afrag(al[i], cur.Al, wm * 32 + i * 16, l, h16);
        }
#pragma unroll
        for (int j = 0; j < 2; ++j) {
            load_bfrag(bh[j], cur.Bh, wn * 32 + j * 16, l, h16);
            load_bfrag(bl[j], cur.Bl, wn * 32 + j * 16, l, h16);
        }
#pragma unroll
        for (int i = 0; i < 2; ++i)
#pragma unroll
            for (int j = 0; j < 2; ++j) {
                acc[i][j] = WMMA_BF16(ah[i].v, bh[j].v, acc[i][j]);
                acc[i][j] = WMMA_BF16(ah[i].v, bl[j].v, acc[i][j]);
                acc[i][j] = WMMA_BF16(al[i].v, bh[j].v, acc[i][j]);
            }
        __syncthreads();                             // done reading 'cur' everywhere
    }

#pragma unroll
    for (int j = 0; j < 2; ++j) {
        int Cg = C0 + wn * 32 + j * 16 + l;
        float s = COLSCALE ? cs[Cg] : 0.25f;
#pragma unroll
        for (int i = 0; i < 2; ++i) {
            int Rg = R0 + wm * 32 + i * 16 + h16 * 8;
#pragma unroll
            for (int r = 0; r < 8; ++r)
                out[(size_t)(Rg + r) * ldo + Cg] = acc[i][j][r] * s;
        }
    }
}

// ---------------------------------------------------------------------------
// Pre-pack kernels (one-shot conversions; keep magic-div VALU out of GEMM loop)
// ---------------------------------------------------------------------------
// Kernel: align-corners bilinear /2 resize of foreground -> fg (B,C,64,64)
__global__ void __launch_bounds__(256) resize_kernel(const float* __restrict__ src,
                                                     float* __restrict__ fg) {
    int idx = blockIdx.x * 256 + threadIdx.x;
    if (idx >= B_ * C_ * Hs * Ws) return;
    int xo = idx & 63, yo = (idx >> 6) & 63, bc = idx >> 12;
    const float* s = src + (size_t)bc * H_ * W_;
    float cy = yo * (127.0f / 63.0f), cx = xo * (127.0f / 63.0f);
    int y0 = (int)floorf(cy), x0 = (int)floorf(cx);
    float wy = cy - y0, wx = cx - x0;
    int y1 = min(y0 + 1, H_ - 1), x1 = min(x0 + 1, W_ - 1);
    float a = s[y0 * W_ + x0], b = s[y0 * W_ + x1];
    float c = s[y1 * W_ + x0], d = s[y1 * W_ + x1];
    fg[idx] = (a * (1.f - wx) + b * wx) * (1.f - wy) + (c * (1.f - wx) + d * wx) * wy;
}

__global__ void __launch_bounds__(256) ss_kernel(const float* __restrict__ fg,
                                                 float* __restrict__ ss) {
    int idx = blockIdx.x * 256 + threadIdx.x;
    if (idx >= B_ * Hs * Ws) return;
    int b = idx >> 12, yx = idx & 4095;
    const float* p = fg + (size_t)b * C_ * Hs * Ws + yx;
    float s = 0.f;
#pragma unroll 8
    for (int c = 0; c < C_; ++c) { float v = p[(size_t)c * Hs * Ws]; s += v * v; }
    ss[idx] = s;
}

// nr[b,p] = SOFTMAX_SCALE / max(||patch p||, 1e-4)   (3x3 box sum of ss)
__global__ void __launch_bounds__(256) nrecip_kernel(const float* __restrict__ ss,
                                                     float* __restrict__ nr) {
    int idx = blockIdx.x * 256 + threadIdx.x;
    if (idx >= B_ * P_) return;
    int b = idx >> 12, p = idx & 4095, py = p >> 6, px = p & 63;
    float s = 0.f;
#pragma unroll
    for (int dy = -1; dy <= 1; ++dy)
#pragma unroll
        for (int dx = -1; dx <= 1; ++dx) {
            int y = py + dy, x = px + dx;
            if ((unsigned)y < (unsigned)Hs && (unsigned)x < (unsigned)Ws)
                s += ss[b * P_ + y * Ws + x];
        }
    nr[idx] = 10.0f / fmaxf(sqrtf(s), 1e-4f);
}

// im2col of fg, bf16 hi/lo packed per row: [hi 576][lo 576] ushorts (2304B/row)
__global__ void __launch_bounds__(256) imc_pack_kernel(const float* __restrict__ fg,
                                                       unsigned short* __restrict__ imc,
                                                       int b) {
    int r = blockIdx.x;              // patch / pixel id
    int py = r >> 6, px = r & 63;
    const float* fgb = fg + (size_t)b * C_ * Hs * Ws;
    for (int k = threadIdx.x; k < K1; k += 256) {
        int cc = k / 9, rem = k - cc * 9, kh = rem / 3, kw = rem - kh * 3;
        int y = py + kh - 1, x = px + kw - 1;
        float v = 0.f;
        if ((unsigned)y < (unsigned)Hs && (unsigned)x < (unsigned)Ws)
            v = fgb[((size_t)cc * Hs + y) * Ws + x];
        unsigned short h = f2bf_hi(v);
        imc[(size_t)r * 1152 + k] = h;
        imc[(size_t)r * 1152 + K1 + k] = f2bf_hi(v - bf2f(h));
    }
}

// BgP transposed [kcol][p], bf16 hi/lo: [hi 4096][lo 4096] ushorts (16KB/row)
__global__ void __launch_bounds__(256) bgp_pack_kernel(const float* __restrict__ bg,
                                                       unsigned short* __restrict__ bgpt,
                                                       int b) {
    int n = blockIdx.x;              // kcol = c*16 + kh*4 + kw
    int cc = n >> 4, kh = (n >> 2) & 3, kw = n & 3;
    const float* bgb = bg + (size_t)b * C_ * H_ * W_;
    for (int p = threadIdx.x; p < P_; p += 256) {
        int y = ((p >> 6) << 1) + kh - 1, x = ((p & 63) << 1) + kw - 1;
        float v = 0.f;
        if ((unsigned)y < (unsigned)H_ && (unsigned)x < (unsigned)W_)
            v = bgb[((size_t)cc * H_ + y) * W_ + x];
        unsigned short h = f2bf_hi(v);
        bgpt[(size_t)n * 8192 + p] = h;
        bgpt[(size_t)n * 8192 + P_ + p] = f2bf_hi(v - bf2f(h));
    }
}

// ---------------------------------------------------------------------------
// Softmax over patch axis (row-contiguous), clip 1e-8, then pack the row
// IN PLACE as bf16 hi/lo planes: row bytes [hi 8192][lo 8192] (was 16384 fp32).
// Reads all precede first barrier; packed writes follow last barrier.
// ---------------------------------------------------------------------------
__global__ void __launch_bounds__(256) softmax_pack_kernel(float* __restrict__ buf) {
    __shared__ float red[256];
    float* p = buf + (size_t)blockIdx.x * P_;
    unsigned short* hi = (unsigned short*)p;
    unsigned short* lo = hi + P_;
    float v[16];
    float mx = -3.4e38f;
#pragma unroll
    for (int i = 0; i < 16; ++i) { v[i] = p[i * 256 + threadIdx.x]; mx = fmaxf(mx, v[i]); }
    red[threadIdx.x] = mx; __syncthreads();
    for (int s = 128; s > 0; s >>= 1) {
        if ((int)threadIdx.x < s) red[threadIdx.x] = fmaxf(red[threadIdx.x], red[threadIdx.x + s]);
        __syncthreads();
    }
    mx = red[0]; __syncthreads();
    float sum = 0.f;
#pragma unroll
    for (int i = 0; i < 16; ++i) { v[i] = __expf(v[i] - mx); sum += v[i]; }
    red[threadIdx.x] = sum; __syncthreads();
    for (int s = 128; s > 0; s >>= 1) {
        if ((int)threadIdx.x < s) red[threadIdx.x] += red[threadIdx.x + s];
        __syncthreads();
    }
    float inv = 1.0f / red[0];
#pragma unroll
    for (int i = 0; i < 16; ++i) {
        int idx = i * 256 + threadIdx.x;
        float w = fmaxf(v[i] * inv, 1e-8f);
        unsigned short h = f2bf_hi(w);
        hi[idx] = h;
        lo[idx] = f2bf_hi(w - bf2f(h));
    }
}

// ---------------------------------------------------------------------------
// col2im for stride-2 transposed conv with flipped 4x4 taps.
// ---------------------------------------------------------------------------
__global__ void __launch_bounds__(256) col2im_kernel(const float* __restrict__ U,
                                                     float* __restrict__ ral, int b) {
    int idx = blockIdx.x * 256 + threadIdx.x;
    if (idx >= C_ * H_ * W_) return;
    int ow = idx & 127, oh = (idx >> 7) & 127, c = idx >> 14;
    float acc = 0.f;
#pragma unroll
    for (int kh = (oh & 1); kh < 4; kh += 2) {
        int t = oh + kh - 2;
        if (t < 0 || t > 126) continue;
        int ih = t >> 1;
#pragma unroll
        for (int kw = (ow & 1); kw < 4; kw += 2) {
            int u = ow + kw - 2;
            if (u < 0 || u > 126) continue;
            int iw = u >> 1;
            acc += U[(size_t)(ih * Ws + iw) * K2 + (c * 16 + (3 - kh) * 4 + (3 - kw))];
        }
    }
    ral[((size_t)b * C_ + c) * H_ * W_ + (idx & (H_ * W_ - 1))] = acc;
}

// ---------------------------------------------------------------------------
// msfa kernels (10% of FLOPs): direct fp32 convs, uniform co => scalar weights
// ---------------------------------------------------------------------------
__global__ void __launch_bounds__(256) msfa_conv1_kernel(const float* __restrict__ x,
                                                         const float* __restrict__ w,
                                                         const float* __restrict__ bias,
                                                         float* __restrict__ h) {
    int tx = threadIdx.x & 15, ty = threadIdx.x >> 4;
    int X = blockIdx.x * 16 + tx, Y = blockIdx.y * 16 + ty;
    int b = blockIdx.z >> 6, co = blockIdx.z & 63;
    const float* xb = x + (size_t)b * C_ * H_ * W_;
    float acc = bias[co];
    for (int ci = 0; ci < C_; ++ci) {
        const float* xc = xb + (size_t)ci * H_ * W_;
        const float* wc = w + ((size_t)co * C_ + ci) * 9;
#pragma unroll
        for (int kh = 0; kh < 3; ++kh) {
            int y = Y + kh - 1;
            if ((unsigned)y >= (unsigned)H_) continue;
#pragma unroll
            for (int kw = 0; kw < 3; ++kw) {
                int xx = X + kw - 1;
                if ((unsigned)xx >= (unsigned)W_) continue;
                acc = fmaf(xc[y * W_ + xx], wc[kh * 3 + kw], acc);
            }
        }
    }
    h[((size_t)b * C_ + co) * H_ * W_ + Y * W_ + X] = fmaxf(acc, 0.f);
}

__global__ void __launch_bounds__(256) wmap_kernel(const float* __restrict__ h,
                                                   const float* __restrict__ w2,
                                                   const float* __restrict__ b2,
                                                   float* __restrict__ wmap) {
    int idx = blockIdx.x * 256 + threadIdx.x;
    if (idx >= B_ * H_ * W_) return;
    int b = idx >> 14, yx = idx & 16383;
    const float* hb = h + (size_t)b * C_ * H_ * W_ + yx;
    float lg[4];
#pragma unroll
    for (int i = 0; i < 4; ++i) {
        float a = b2[i];
        for (int c = 0; c < C_; ++c) a = fmaf(hb[(size_t)c * H_ * W_], w2[i * C_ + c], a);
        lg[i] = fmaxf(a, 0.f);
    }
    float mx = fmaxf(fmaxf(lg[0], lg[1]), fmaxf(lg[2], lg[3]));
    float e[4], sum = 0.f;
#pragma unroll
    for (int i = 0; i < 4; ++i) { e[i] = __expf(lg[i] - mx); sum += e[i]; }
    float inv = 1.0f / sum;
#pragma unroll
    for (int i = 0; i < 4; ++i)
        wmap[((size_t)b * 4 + i) * H_ * W_ + yx] = e[i] * inv;
}

__global__ void __launch_bounds__(256) msfa_out_kernel(const float* __restrict__ x,
                                                       const float* __restrict__ dw,
                                                       const float* __restrict__ db,
                                                       const float* __restrict__ wmap,
                                                       float* __restrict__ out) {
    int tx = threadIdx.x & 15, ty = threadIdx.x >> 4;
    int X = blockIdx.x * 16 + tx, Y = blockIdx.y * 16 + ty;
    int b = blockIdx.z >> 6, co = blockIdx.z & 63;
    const float* xb = x + (size_t)b * C_ * H_ * W_;
    int yxo = Y * W_ + X;
    float res = 0.f;
#pragma unroll
    for (int i = 0; i < 4; ++i) {
        int d = 1 << i;
        float acc = db[i * C_ + co];
        const float* wi = dw + ((size_t)i * C_ + co) * C_ * 9;
        for (int ci = 0; ci < C_; ++ci) {
            const float* xc = xb + (size_t)ci * H_ * W_;
            const float* wc = wi + ci * 9;
#pragma unroll
            for (int kh = 0; kh < 3; ++kh) {
                int y = Y + (kh - 1) * d;
                if ((unsigned)y >= (unsigned)H_) continue;
#pragma unroll
                for (int kw = 0; kw < 3; ++kw) {
                    int xx = X + (kw - 1) * d;
                    if ((unsigned)xx >= (unsigned)W_) continue;
                    acc = fmaf(xc[y * W_ + xx], wc[kh * 3 + kw], acc);
                }
            }
        }
        res += wmap[((size_t)b * 4 + i) * H_ * W_ + yxo] * fmaxf(acc, 0.f);
    }
    out[((size_t)b * C_ + co) * H_ * W_ + yxo] = res;
}

// ---------------------------------------------------------------------------
extern "C" void kernel_launch(void* const* d_in, const int* in_sizes, int n_in,
                              void* d_out, int out_size, void* d_ws, size_t ws_size,
                              hipStream_t stream) {
    const float* background = (const float*)d_in[0];
    const float* foreground = (const float*)d_in[1];
    const float* dil_w = (const float*)d_in[2];
    const float* dil_b = (const float*)d_in[3];
    const float* wc1_w = (const float*)d_in[4];
    const float* wc1_b = (const float*)d_in[5];
    const float* wc2_w = (const float*)d_in[6];
    const float* wc2_b = (const float*)d_in[7];
    float* out = (float*)d_out;
    (void)in_sizes; (void)n_in; (void)out_size; (void)ws_size;

    char* ws = (char*)d_ws;
    size_t off = 0;
    auto wsa = [&](size_t bytes) -> void* {
        void* p = ws + off;
        off += (bytes + 255) & ~(size_t)255;
        return p;
    };
    float* fg    = (float*)wsa((size_t)B_ * C_ * Hs * Ws * 4);     //  4 MB
    float* ss    = (float*)wsa((size_t)B_ * Hs * Ws * 4);          // 64 KB
    float* nr    = (float*)wsa((size_t)B_ * P_ * 4);               // 64 KB
    float* ral   = (float*)wsa((size_t)B_ * C_ * H_ * W_ * 4);     // 16 MB
    float* Ubuf  = (float*)wsa((size_t)P_ * K2 * 4);               // 16 MB (per batch)
    unsigned short* imc  = (unsigned short*)wsa((size_t)P_ * 1152 * 2);  // 9.4 MB
    unsigned short* bgpt = (unsigned short*)wsa((size_t)K2 * 8192 * 2);  // 16 MB
    float* big   = (float*)wsa((size_t)P_ * P_ * 4);               // 64 MB (score->attn)
    float* hbuf  = big;                                            // reuse after ral loop
    float* wmapb = big + (size_t)B_ * C_ * H_ * W_;

    resize_kernel<<<(B_ * C_ * Hs * Ws + 255) / 256, 256, 0, stream>>>(foreground, fg);
    ss_kernel<<<(B_ * Hs * Ws + 255) / 256, 256, 0, stream>>>(fg, ss);
    nrecip_kernel<<<(B_ * P_ + 255) / 256, 256, 0, stream>>>(ss, nr);

    for (int b = 0; b < B_; ++b) {
        imc_pack_kernel<<<P_, 256, 0, stream>>>(fg, imc, b);
        bgp_pack_kernel<<<K2, 256, 0, stream>>>(background, bgpt, b);
        dim3 g1(P_ / BM, P_ / BN);          // 32 x 64
        gemm_kernel<K1, true><<<g1, 256, 0, stream>>>(
            (const char*)imc, 2304, (const char*)imc, 2304, big, P_, nr + b * P_);
        softmax_pack_kernel<<<P_, 256, 0, stream>>>(big);
        dim3 g2(P_ / BM, K2 / BN);          // 32 x 16
        gemm_kernel<P_, false><<<g2, 256, 0, stream>>>(
            (const char*)big, 16384, (const char*)bgpt, 16384, Ubuf, K2, nullptr);
        col2im_kernel<<<(C_ * H_ * W_ + 255) / 256, 256, 0, stream>>>(Ubuf, ral, b);
    }

    dim3 cg(W_ / 16, H_ / 16, B_ * C_);
    msfa_conv1_kernel<<<cg, 256, 0, stream>>>(ral, wc1_w, wc1_b, hbuf);
    wmap_kernel<<<(B_ * H_ * W_ + 255) / 256, 256, 0, stream>>>(hbuf, wc2_w, wc2_b, wmapb);
    msfa_out_kernel<<<cg, 256, 0, stream>>>(ral, dil_w, dil_b, wmapb, out);
}
